// SelfAttention_29334626632251
// MI455X (gfx1250) — matrix operand order repaired and checked by reading, not compiled, not run on hardware
//
#include <hip/hip_runtime.h>

// ---------------------------------------------------------------------------
// Self-attention forward on gfx1250 (MI455X).
// fp32 -> bf16, matmuls via v_wmma_f32_16x16x32_bf16 (fp32 accum).
// Flash attention computed TRANSPOSED (S^T = K Q^T, O^T = V^T P^T) so softmax
// stats per query are lane-local (one xor-16 shuffle instead of 4-step trees).
// GEMMs register-double-buffer fragment loads to overlap VMEM with WMMA.
// ---------------------------------------------------------------------------

typedef __attribute__((ext_vector_type(16))) __bf16 v16bf;
typedef __attribute__((ext_vector_type(8)))  __bf16 v8bf;
typedef __attribute__((ext_vector_type(8)))  float  v8f;

__device__ __forceinline__ __bf16 f2bf(float f) {
    union { float f; unsigned u; } x; x.f = f;
    unsigned r = x.u + 0x7FFFu + ((x.u >> 16) & 1u);   // round-to-nearest-even
    unsigned short h = (unsigned short)(r >> 16);
    __bf16 out;
    __builtin_memcpy(&out, &h, 2);
    return out;
}

// A-fragment (16x32 bf16, M x K), row-major source, lda in elements.
// Lane l: row m0+(l&15); low lanes K=k0+{0..7,16..23}, high lanes +8.
__device__ __forceinline__ v16bf load_frag_a(const __bf16* base, int lda, int m0, int k0) {
    int lane = threadIdx.x & 31;
    int m  = m0 + (lane & 15);
    int kb = k0 + ((lane & 16) >> 1);
    const __bf16* p = base + (size_t)m * lda + kb;
    v16bf r;
    ((v8bf*)&r)[0] = *(const v8bf*)(p);
    ((v8bf*)&r)[1] = *(const v8bf*)(p + 16);
    return r;
}

// B-fragment (32x16 bf16, K x N) from column-major B stored as Bt[N][K].
// Lane l: col n0+(l&15); low lanes K=k0+{0..7,16..23}, high lanes +8.
__device__ __forceinline__ v16bf load_frag_b(const __bf16* baseT, int ldb, int n0, int k0) {
    int lane = threadIdx.x & 31;
    int n  = n0 + (lane & 15);
    int kb = k0 + ((lane & 16) >> 1);
    const __bf16* p = baseT + (size_t)n * ldb + kb;
    v16bf r;
    ((v8bf*)&r)[0] = *(const v8bf*)(p);
    ((v8bf*)&r)[1] = *(const v8bf*)(p + 16);
    return r;
}

__device__ __forceinline__ v8f wmma_bf16(v16bf a, v16bf b, v8f c) {
    return __builtin_amdgcn_wmma_f32_16x16x32_bf16(
        false, a, false, b, (short)0, c, false, false);
}

// ---------------------------------------------------------------------------
// Elementwise prep kernels
// ---------------------------------------------------------------------------
__global__ void k_f32_to_bf16(const float* __restrict__ src, __bf16* __restrict__ dst, int n) {
    int i = blockIdx.x * blockDim.x + threadIdx.x;
    if (i < n) dst[i] = f2bf(src[i]);
}

// src [rows][cols] fp32 -> dst [cols][rows] bf16
__global__ void k_transpose_to_bf16(const float* __restrict__ src, __bf16* __restrict__ dst,
                                    int rows, int cols) {
    int i = blockIdx.x * blockDim.x + threadIdx.x;
    if (i < rows * cols) {
        int r = i / cols, c = i % cols;
        dst[(size_t)c * rows + r] = f2bf(src[i]);
    }
}

// vT[b][h][d][s] = qkv_bf16[(b*S+s)*3072 + 2048 + h*64 + d]
__global__ void k_build_vT(const __bf16* __restrict__ qkv, __bf16* __restrict__ vT) {
    const int S = 2048, DH = 64, H = 16, B = 2;
    int i = blockIdx.x * blockDim.x + threadIdx.x;
    if (i < B * H * DH * S) {
        int s = i % S; int t = i / S;
        int d = t % DH; t /= DH;
        int h = t % H;  int b = t / H;
        vT[i] = qkv[((size_t)(b * S + s)) * 3072 + 2048 + h * DH + d];
    }
}

// ---------------------------------------------------------------------------
// GEMM: C[M,N] = A[M,K](bf16 row-major) @ B (given as Bt[N,K] bf16) + bias.
// 256 threads = 8 waves, block tile 128x128, wave tile 32x64.
// Register double-buffered fragment loads (overlap VMEM with WMMA).
// ---------------------------------------------------------------------------
template <bool OUT_BF16>
__global__ void k_gemm_bf16(const __bf16* __restrict__ A, const __bf16* __restrict__ Bt,
                            const float* __restrict__ bias, void* __restrict__ C,
                            int M, int N, int K) {
    int wave = threadIdx.x >> 5;
    int lane = threadIdx.x & 31;
    int bm = blockIdx.y * 128 + (wave >> 1) * 32;
    int bn = blockIdx.x * 128 + (wave & 1) * 64;

    v8f acc[2][4] = {};
    v16bf af[2][2], bf[2][4];

    auto load_tile = [&](int t, int buf) {
        int k0 = t << 5;
        af[buf][0] = load_frag_a(A, K, bm, k0);
        af[buf][1] = load_frag_a(A, K, bm + 16, k0);
#pragma unroll
        for (int j = 0; j < 4; ++j) bf[buf][j] = load_frag_b(Bt, K, bn + 16 * j, k0);
    };

    int nk = K >> 5;
    load_tile(0, 0);
    for (int t = 0; t < nk; ++t) {
        int cur = t & 1;
        if (t + 1 < nk) load_tile(t + 1, cur ^ 1);
#pragma unroll
        for (int i = 0; i < 2; ++i)
#pragma unroll
            for (int j = 0; j < 4; ++j)
                acc[i][j] = wmma_bf16(af[cur][i], bf[cur][j], acc[i][j]);
    }

    int nl = lane & 15;
    int mh = (lane & 16) >> 1;
#pragma unroll
    for (int i = 0; i < 2; ++i) {
#pragma unroll
        for (int j = 0; j < 4; ++j) {
            int n = bn + 16 * j + nl;
            float bv = bias[n];
#pragma unroll
            for (int r = 0; r < 8; ++r) {
                int m = bm + 16 * i + r + mh;
                float v = acc[i][j][r] + bv;
                if (OUT_BF16) ((__bf16*)C)[(size_t)m * N + n] = f2bf(v);
                else          ((float*) C)[(size_t)m * N + n] = v;
            }
        }
    }
}

// ---------------------------------------------------------------------------
// Transposed flash attention. Grid: (S/128, H, B); 256 threads = 8 waves.
// Each wave owns 16 QUERIES (columns of all fragments); streams keys 32/block.
//   S^T = K @ Q^T   (A = K rows, B = Q rows as column-major Q^T)
//   O^T = V^T @ P^T (A = vT rows, B = P^T staged through LDS as Bt = P)
// Per-query softmax stats are lane-local scalars + ONE xor-16 shuffle.
// ---------------------------------------------------------------------------
__global__ void k_flash_attn(const __bf16* __restrict__ qkv, const __bf16* __restrict__ vT,
                             __bf16* __restrict__ ctx) {
    const int S = 2048, D = 1024, DH = 64, LDQ = 3 * D;
    int b = blockIdx.z, h = blockIdx.y;
    int wave = threadIdx.x >> 5;
    int lane = threadIdx.x & 31;
    int nl = lane & 15;                 // query within wave tile / B-frag column
    int hi = (lane & 16) >> 4;          // fragment half
    int q0 = blockIdx.x * 128 + wave * 16;

    const __bf16* Qb = qkv + ((size_t)b * S) * LDQ + h * DH;
    const __bf16* Kb = qkv + ((size_t)b * S) * LDQ + D + h * DH;
    const __bf16* Vt = vT + (((size_t)b * 16 + h) * DH) * S;    // [d][s], ld=S

    __shared__ __attribute__((aligned(16))) __bf16 pshare[8][16 * 32]; // [q][k] per wave
    __bf16* pw = &pshare[wave][0];

    v8f o[4] = {};                      // O^T: 64(d) x 16(q), 4 M-tiles
    float mcol = -1e30f, lcol = 0.f;    // per-query (per-lane) running stats

    // Q^T B-fragments (d = K-dim), reused for all key blocks
    v16bf qbf[2];
    qbf[0] = load_frag_b(Qb, LDQ, q0, 0);
    qbf[1] = load_frag_b(Qb, LDQ, q0, 32);

    for (int s0 = 0; s0 < S; s0 += 32) {
        if (s0 + 32 < S) {  // prefetch next key/value block (global_prefetch_b8)
            __builtin_prefetch((const void*)(Kb + (size_t)(s0 + 32 + lane) * LDQ), 0, 3);
            __builtin_prefetch((const void*)(Vt + (size_t)lane * S + s0 + 32), 0, 3);
            __builtin_prefetch((const void*)(Vt + (size_t)(lane + 32) * S + s0 + 32), 0, 3);
        }

        // ---- S^T: 32(keys) x 16(queries), K-dim = d = 64 ----
        v8f st[2] = {};
#pragma unroll
        for (int i = 0; i < 2; ++i) {
            v16bf ka0 = load_frag_a(Kb, LDQ, s0 + 16 * i, 0);
            v16bf ka1 = load_frag_a(Kb, LDQ, s0 + 16 * i, 32);
            st[i] = wmma_bf16(ka0, qbf[0], st[i]);
            st[i] = wmma_bf16(ka1, qbf[1], st[i]);
        }

        // ---- online softmax over keys (per query = per lane column) ----
        float lm = -1e30f;
#pragma unroll
        for (int i = 0; i < 2; ++i)
#pragma unroll
            for (int r = 0; r < 8; ++r) {
                float v = st[i][r] * 0.125f;    // 1/sqrt(64)
                st[i][r] = v;
                lm = fmaxf(lm, v);
            }
        lm = fmaxf(lm, __shfl_xor(lm, 16, 32));     // other 16 keys live in lane^16

        float mn = fmaxf(mcol, lm);
        float alpha = __expf(mcol - mn);
        mcol = mn;

        float psum = 0.f;
#pragma unroll
        for (int i = 0; i < 2; ++i)
#pragma unroll
            for (int r = 0; r < 8; ++r) {
                float e = __expf(st[i][r] - mn);
                st[i][r] = e;
                psum += e;
            }
        psum += __shfl_xor(psum, 16, 32);
        lcol = lcol * alpha + psum;

        // rescale running output (alpha uniform within lane)
#pragma unroll
        for (int j = 0; j < 4; ++j)
#pragma unroll
            for (int r = 0; r < 8; ++r) o[j][r] *= alpha;

        // ---- P^T (C-frag, rows=keys, cols=queries) -> LDS as P[q][k] ----
#pragma unroll
        for (int i = 0; i < 2; ++i)
#pragma unroll
            for (int r = 0; r < 8; ++r) {
                int k = r + 8 * hi + 16 * i;
                pw[nl * 32 + k] = f2bf(st[i][r]);
            }
        asm volatile("s_wait_dscnt 0" ::: "memory");
        v16bf pbf = load_frag_b(pw, 32, 0, 0);      // B-frag of P^T (Bt = P)

        // ---- O^T += V^T(64 x 32keys) @ P^T(32keys x 16q) ----
#pragma unroll
        for (int j = 0; j < 4; ++j) {
            v16bf va = load_frag_a(Vt, S, 16 * j, s0);
            o[j] = wmma_bf16(va, pbf, o[j]);
        }
    }

    // ---- normalize and write context (bf16, [B,S,D]); divisor lane-uniform ----
    float rcl = 1.0f / lcol;
    int q = q0 + nl;
    __bf16* crow = ctx + ((size_t)b * S + q) * D + h * DH;
#pragma unroll
    for (int j = 0; j < 4; ++j)
#pragma unroll
        for (int r = 0; r < 8; ++r)
            crow[16 * j + r + 8 * hi] = f2bf(o[j][r] * rcl);
}

// ---------------------------------------------------------------------------
// Launch
// ---------------------------------------------------------------------------
extern "C" void kernel_launch(void* const* d_in, const int* in_sizes, int n_in,
                              void* d_out, int out_size, void* d_ws, size_t ws_size,
                              hipStream_t stream) {
    const int B = 2, S = 2048, D = 1024, H = 16, DH = 64;
    const int M = B * S;        // 4096
    const int N3 = 3 * D;       // 3072

    const float* x     = (const float*)d_in[0];
    const float* w_in  = (const float*)d_in[1];
    const float* b_in  = (const float*)d_in[2];
    const float* w_out = (const float*)d_in[3];
    const float* b_out = (const float*)d_in[4];
    float* out = (float*)d_out;

    char* ws = (char*)d_ws;
    size_t off = 0;
    __bf16* x_bf   = (__bf16*)(ws + off); off += (size_t)M * D * 2;          // 8 MiB
    __bf16* win_t  = (__bf16*)(ws + off); off += (size_t)N3 * D * 2;         // 6 MiB
    __bf16* wout_t = (__bf16*)(ws + off); off += (size_t)D * D * 2;          // 2 MiB
    __bf16* qkv    = (__bf16*)(ws + off); off += (size_t)M * N3 * 2;         // 24 MiB
    __bf16* vT     = (__bf16*)(ws + off); off += (size_t)B * H * DH * S * 2; // 8 MiB
    __bf16* ctx    = (__bf16*)(ws + off); off += (size_t)M * D * 2;          // 8 MiB
    (void)ws_size; (void)in_sizes; (void)n_in; (void)out_size;

    {
        int n = M * D;
        k_f32_to_bf16<<<(n + 255) / 256, 256, 0, stream>>>(x, x_bf, n);
    }
    {
        int n = D * N3;
        k_transpose_to_bf16<<<(n + 255) / 256, 256, 0, stream>>>(w_in, win_t, D, N3);
    }
    {
        int n = D * D;
        k_transpose_to_bf16<<<(n + 255) / 256, 256, 0, stream>>>(w_out, wout_t, D, D);
    }

    // QKV projection: qkv[M,3D] = x @ w_in + b_in (bf16 out)
    k_gemm_bf16<true><<<dim3(N3 / 128, M / 128), 256, 0, stream>>>(
        x_bf, win_t, b_in, (void*)qkv, M, N3, D);

    // V transpose per head for contiguous PV fragments
    {
        int n = B * H * DH * S;
        k_build_vT<<<(n + 255) / 256, 256, 0, stream>>>(qkv, vT);
    }

    // Flash attention -> ctx (bf16, [B,S,D])
    k_flash_attn<<<dim3(S / 128, H, B), 256, 0, stream>>>(qkv, vT, ctx);

    // Output projection: out[M,D] = ctx @ w_out + b_out (fp32 out)
    k_gemm_bf16<false><<<dim3(D / 128, M / 128), 256, 0, stream>>>(
        ctx, wout_t, b_out, (void*)out, M, D, D);
}
